// IGMSA_661424964051
// MI455X (gfx1250) — matrix operand order repaired and checked by reading, not compiled, not run on hardware
//
#include <hip/hip_runtime.h>
#include <math.h>

// ---------------------------------------------------------------------------
// Types / helpers
// ---------------------------------------------------------------------------
typedef __attribute__((ext_vector_type(16))) __bf16 bf16x16;
typedef __attribute__((ext_vector_type(8)))  float  f32x8;

struct Frag {
    union {
        bf16x16        v;
        unsigned short u[16];
        unsigned int   d[8];
        uint4          q[2];
    };
};

static __device__ __forceinline__ unsigned short f2bf(float f) {
    union { float f; unsigned u; } a; a.f = f;
    unsigned r = a.u + 0x7FFFu + ((a.u >> 16) & 1u);
    return (unsigned short)(r >> 16);
}
static __device__ __forceinline__ float bf2f(unsigned short h) {
    union { unsigned u; float f; } a; a.u = ((unsigned)h) << 16; return a.f;
}
static __device__ __forceinline__ f32x8 zero8() {
    f32x8 z;
#pragma unroll
    for (int i = 0; i < 8; ++i) z[i] = 0.0f;
    return z;
}
static __device__ __forceinline__ f32x8 wmma_bf16(const Frag& A, const Frag& B, f32x8 C) {
    return __builtin_amdgcn_wmma_f32_16x16x32_bf16(
        /*neg_a=*/false, A.v, /*neg_b=*/false, B.v,
        /*c_mod=*/(short)0, C, /*reuse_a=*/false, /*reuse_b=*/false);
}

// CDNA5 async global->LDS copy (ASYNCcnt-tracked DMA, no VGPR round trip).
// vdst operand carries the LDS byte offset (low 32 bits of the generic
// shared-memory pointer, per the LDS aperture mapping).
static __device__ __forceinline__ void async_g2l_b64(void* lds_ptr, const void* gptr) {
    unsigned lds_off = (unsigned)(unsigned long long)lds_ptr;
    asm volatile("global_load_async_to_lds_b64 %0, %1, off"
                 :: "v"(lds_off), "v"(gptr)
                 : "memory");
}
static __device__ __forceinline__ void wait_asynccnt0() {
    asm volatile("s_wait_asynccnt 0x0" ::: "memory");
}

// Problem constants
#define B_   4
#define HW_  256
#define N_   65536       // 256*256
#define C_   128
#define HEADS_ 4
#define DH_  32

// ---------------------------------------------------------------------------
// K0: transpose Wq/Wk/Wv -> bf16 [out][in], and zero Gram/sumsq accumulators
// ---------------------------------------------------------------------------
__global__ void k0_prep(const float* __restrict__ Wq, const float* __restrict__ Wk,
                        const float* __restrict__ Wv,
                        unsigned short* __restrict__ Wtq, unsigned short* __restrict__ Wtk,
                        unsigned short* __restrict__ Wtv,
                        float* __restrict__ zero_base /* Gram+ssq+ssk, contiguous */) {
    int idx = blockIdx.x * 256 + threadIdx.x;
    if (idx < 3 * 16384) {
        int m = idx / 16384;
        int r = (idx % 16384) / 128;   // input channel
        int c = idx % 128;             // output channel
        const float* W = (m == 0) ? Wq : (m == 1) ? Wk : Wv;
        unsigned short* Wt = (m == 0) ? Wtq : (m == 1) ? Wtk : Wtv;
        Wt[c * 128 + r] = f2bf(W[r * 128 + c]);
    } else {
        int z = idx - 3 * 16384;
        if (z < 17408) zero_base[z] = 0.0f;   // 16384 Gram + 512 ssq + 512 ssk
    }
}

// ---------------------------------------------------------------------------
// K1: projections q,k (bf16) and v (f32).  One block = 128 rows of X.
// ---------------------------------------------------------------------------
__global__ void k1_proj(const float* __restrict__ x,
                        const unsigned short* __restrict__ Wtq,
                        const unsigned short* __restrict__ Wtk,
                        const unsigned short* __restrict__ Wtv,
                        unsigned short* __restrict__ qws,
                        unsigned short* __restrict__ kws,
                        float* __restrict__ vws) {
    __shared__ unsigned short As[128 * 144];   // 128 rows x 128 cols bf16, padded
    const int tid  = threadIdx.x;
    const int lane = tid & 31;
    const int half = lane >> 4;
    const int l15  = lane & 15;
    const long row0 = (long)blockIdx.x * 128;

    // prefetch next block's X tile while we work on this one
    if (blockIdx.x + 1 < 2048)
        __builtin_prefetch(x + (row0 + 128) * 128 + tid * 64, 0, 3);

    // stage X tile as bf16 into LDS (coalesced float4 loads)
#pragma unroll
    for (int i = 0; i < 16; ++i) {
        int e = (tid + 256 * i) * 4;
        float4 xv = *(const float4*)(x + row0 * 128 + e);
        int r = e >> 7, c = e & 127;
        unsigned short* p = &As[r * 144 + c];
        p[0] = f2bf(xv.x); p[1] = f2bf(xv.y); p[2] = f2bf(xv.z); p[3] = f2bf(xv.w);
    }
    __syncthreads();

    // 8 row-strips x 3 matrices x 8 col-tiles = 192 jobs over 8 waves.
    // Job decode is wave-uniform: force scalarization via readfirstlane so the
    // store-path select compiles to s_cbranch instead of exec-mask ladders.
    for (int job = tid >> 5; job < 192; job += 8) {
        const int js    = __builtin_amdgcn_readfirstlane(job);
        const int strip = js / 24;
        const int rem   = js % 24;
        const int m     = rem >> 3;
        const int ct    = rem & 7;
        const unsigned short* Wt = (m == 0) ? Wtq : (m == 1) ? Wtk : Wtv;
        const int col  = ct * 16 + l15;
        const int lrow = strip * 16 + l15;

        f32x8 acc = zero8();
#pragma unroll
        for (int kk = 0; kk < 4; ++kk) {
            int kb = kk * 32;
            Frag a, b;
            a.q[0] = *(const uint4*)(&As[lrow * 144 + kb + half * 16]);
            a.q[1] = *(const uint4*)(&As[lrow * 144 + kb + half * 16 + 8]);
            const uint4* pb = (const uint4*)(Wt + col * 128 + kb + half * 16);
            b.q[0] = pb[0];
            b.q[1] = pb[1];
            acc = wmma_bf16(a, b, acc);
        }
        const long grow = row0 + strip * 16;
        if (m == 2) {
#pragma unroll
            for (int r = 0; r < 8; ++r) {
                int orow = half ? (r + 8) : r;
                vws[(grow + orow) * 128 + col] = acc[r];
            }
        } else {
            unsigned short* dst = m ? kws : qws;
#pragma unroll
            for (int r = 0; r < 8; ++r) {
                int orow = half ? (r + 8) : r;
                dst[(grow + orow) * 128 + col] = f2bf(acc[r]);
            }
        }
    }
}

// ---------------------------------------------------------------------------
// K2: per (b,head,chunk) partial Gram G = K^T Q (32x32) via WMMA + sumsq.
//     grid.x = 16 * 128 ; chunk = 512 rows of n.
//     LDS staging uses CDNA5 async global->LDS DMA (ASYNCcnt).
// ---------------------------------------------------------------------------
__global__ void k2_gram(const unsigned short* __restrict__ qws,
                        const unsigned short* __restrict__ kws,
                        float* __restrict__ Gram,
                        float* __restrict__ ssq,
                        float* __restrict__ ssk) {
    __shared__ unsigned short qb[512 * 36];
    __shared__ unsigned short kb[512 * 36];
    const int tid  = threadIdx.x;
    const int lane = tid & 31;
    const int wv   = tid >> 5;
    const int half = lane >> 4;
    const int l15  = lane & 15;
    const int bh    = blockIdx.x >> 7;
    const int chunk = blockIdx.x & 127;
    const int h     = bh & 3;
    const long rowbase = (long)(bh >> 2) * N_ + (long)chunk * 512;

    // async DMA: 8-byte strips per lane, 16 iterations x 2 matrices
#pragma unroll
    for (int i = 0; i < 16; ++i) {
        int qd = tid + 256 * i;
        int r  = qd >> 3;
        int cq = qd & 7;
        long ga = (rowbase + r) * 128 + h * 32 + cq * 4;
        async_g2l_b64(&qb[r * 36 + cq * 4], qws + ga);
        async_g2l_b64(&kb[r * 36 + cq * 4], kws + ga);
    }
    wait_asynccnt0();
    __syncthreads();

    // column sum-of-squares (for L2 normalization factors)
    {
        int c = tid & 31, rg = tid >> 5;
        float sq = 0.f, sk = 0.f;
        for (int k = 0; k < 64; ++k) {
            int r = rg + 8 * k;
            float qv = bf2f(qb[r * 36 + c]);
            float kv = bf2f(kb[r * 36 + c]);
            sq += qv * qv;
            sk += kv * kv;
        }
        atomicAdd(&ssq[bh * 32 + c], sq);
        atomicAdd(&ssk[bh * 32 + c], sk);
    }

    // Gram via WMMA: A = K^T (transposed LDS reads), B = Q
    f32x8 acc[2][2];
#pragma unroll
    for (int dt = 0; dt < 2; ++dt)
#pragma unroll
        for (int et = 0; et < 2; ++et) acc[dt][et] = zero8();

    const int lr0 = wv * 64;
#pragma unroll
    for (int kk = 0; kk < 2; ++kk) {
        int kb0 = lr0 + kk * 32;
        Frag A[2], Bf[2];
#pragma unroll
        for (int dt = 0; dt < 2; ++dt) {
            int dd = dt * 16 + l15;
#pragma unroll
            for (int j = 0; j < 8; ++j) {
                A[dt].u[j]     = kb[(kb0 + (half ? 8 : 0) + j) * 36 + dd];
                A[dt].u[j + 8] = kb[(kb0 + 16 + (half ? 8 : 0) + j) * 36 + dd];
            }
        }
#pragma unroll
        for (int et = 0; et < 2; ++et) {
            int ee = et * 16 + l15;
#pragma unroll
            for (int j = 0; j < 8; ++j) {
                Bf[et].u[j]     = qb[(kb0 + half * 8 + j) * 36 + ee];
                Bf[et].u[j + 8] = qb[(kb0 + 16 + half * 8 + j) * 36 + ee];
            }
        }
#pragma unroll
        for (int dt = 0; dt < 2; ++dt)
#pragma unroll
            for (int et = 0; et < 2; ++et)
                acc[dt][et] = wmma_bf16(A[dt], Bf[et], acc[dt][et]);
    }

    float* Gb = Gram + bh * 1024;
#pragma unroll
    for (int dt = 0; dt < 2; ++dt)
#pragma unroll
        for (int et = 0; et < 2; ++et)
#pragma unroll
            for (int r = 0; r < 8; ++r) {
                int dd = dt * 16 + (half ? (r + 8) : r);
                int ee = et * 16 + l15;
                atomicAdd(&Gb[dd * 32 + ee], acc[dt][et][r]);
            }
}

// ---------------------------------------------------------------------------
// K3: attn = softmax(G * invk * invq * rescale) ; Mt[c][(h,e)] = sum_d attn[d,e]*Wp[(h,d),c]
// ---------------------------------------------------------------------------
__global__ void k3_attn(const float* __restrict__ Gram, const float* __restrict__ ssq,
                        const float* __restrict__ ssk, const float* __restrict__ rescale,
                        const float* __restrict__ Wp, unsigned short* __restrict__ Mt) {
    __shared__ float G[1024];
    __shared__ float at[32 * 33];
    __shared__ float invq[32], invk[32];
    const int tid = threadIdx.x;
    const int bh  = blockIdx.x;
    const int b   = bh >> 2;
    const int h   = bh & 3;
#pragma unroll
    for (int i = 0; i < 4; ++i) G[tid + 256 * i] = Gram[bh * 1024 + tid + 256 * i];
    if (tid < 32) {
        invq[tid] = 1.0f / fmaxf(sqrtf(ssq[bh * 32 + tid]), 1e-12f);
        invk[tid] = 1.0f / fmaxf(sqrtf(ssk[bh * 32 + tid]), 1e-12f);
    }
    __syncthreads();
    if (tid < 32) {
        const int d = tid;
        const float rs = rescale[h];
        float s[32], m = -3.4e38f;
#pragma unroll
        for (int e = 0; e < 32; ++e) {
            float v = G[d * 32 + e] * invk[d] * invq[e] * rs;
            s[e] = v;
            m = fmaxf(m, v);
        }
        float sum = 0.f;
#pragma unroll
        for (int e = 0; e < 32; ++e) { float p = __expf(s[e] - m); at[d * 33 + e] = p; sum += p; }
        float inv = 1.0f / sum;
#pragma unroll
        for (int e = 0; e < 32; ++e) at[d * 33 + e] *= inv;
    }
    __syncthreads();
#pragma unroll
    for (int i = 0; i < 16; ++i) {
        int idx = tid + 256 * i;
        int c = idx >> 5, e = idx & 31;
        float sum = 0.f;
#pragma unroll
        for (int d = 0; d < 32; ++d) sum += at[d * 33 + e] * Wp[(h * 32 + d) * 128 + c];
        Mt[b * 16384 + c * 128 + h * 32 + e] = f2bf(sum);
    }
}

// ---------------------------------------------------------------------------
// K4: fused dwconv3x3 -> exact GELU -> dwconv3x3 on v_inp, writes base of out.
//     One block = 8x8 spatial tile, all 128 channels.
// ---------------------------------------------------------------------------
__global__ void k4_conv(const float* __restrict__ v, const float* __restrict__ w1,
                        const float* __restrict__ w2, float* __restrict__ out) {
    __shared__ float sin_[12 * 12 * 128];
    __shared__ float smid[10 * 10 * 128];
    const int tid = threadIdx.x;
    const int b   = blockIdx.x >> 10;
    const int t   = blockIdx.x & 1023;
    const int y0  = (t >> 5) * 8;
    const int x0  = (t & 31) * 8;

#pragma unroll 4
    for (int i = 0; i < 72; ++i) {
        int idx = tid + 256 * i;
        int pos = idx >> 7, c = idx & 127;
        int iy = pos / 12, ix = pos % 12;
        int gy = y0 - 2 + iy, gx = x0 - 2 + ix;
        float val = 0.f;
        if (gy >= 0 && gy < HW_ && gx >= 0 && gx < HW_)
            val = v[(((long)b * HW_ + gy) * HW_ + gx) * 128 + c];
        sin_[idx] = val;
    }
    __syncthreads();
#pragma unroll 2
    for (int i = 0; i < 50; ++i) {
        int idx = tid + 256 * i;
        int pos = idx >> 7, c = idx & 127;
        int my = pos / 10, mx = pos % 10;
        float sum = 0.f;
#pragma unroll
        for (int dy = 0; dy < 3; ++dy)
#pragma unroll
            for (int dx = 0; dx < 3; ++dx)
                sum += sin_[((my + dy) * 12 + mx + dx) * 128 + c] * w1[(dy * 3 + dx) * 128 + c];
        smid[idx] = 0.5f * sum * (1.0f + erff(sum * 0.70710678118f));  // exact GELU
    }
    __syncthreads();
#pragma unroll 2
    for (int i = 0; i < 32; ++i) {
        int idx = tid + 256 * i;
        int pos = idx >> 7, c = idx & 127;
        int oy = pos >> 3, ox = pos & 7;
        float sum = 0.f;
#pragma unroll
        for (int dy = 0; dy < 3; ++dy)
#pragma unroll
            for (int dx = 0; dx < 3; ++dx)
                sum += smid[((oy + dy) * 10 + ox + dx) * 128 + c] * w2[(dy * 3 + dx) * 128 + c];
        out[(((long)b * HW_ + y0 + oy) * HW_ + x0 + ox) * 128 + c] = sum;
    }
}

// ---------------------------------------------------------------------------
// K5: x1 = (v * ilf) @ M_b + bp, accumulated onto conv result in out (WMMA).
// ---------------------------------------------------------------------------
__global__ void k5_out(const float* __restrict__ vws, const float* __restrict__ ilf,
                       const unsigned short* __restrict__ Mt, const float* __restrict__ bp,
                       float* __restrict__ out) {
    __shared__ unsigned short As[128 * 144];
    const int tid  = threadIdx.x;
    const int lane = tid & 31;
    const int half = lane >> 4;
    const int l15  = lane & 15;
    const long row0 = (long)blockIdx.x * 128;
    const unsigned short* Mtb = Mt + (row0 >> 16) * 16384;   // per-batch composed matrix

    if (blockIdx.x + 1 < 2048) {
        __builtin_prefetch(vws + (row0 + 128) * 128 + tid * 64, 0, 3);
        __builtin_prefetch(ilf + (row0 + 128) * 128 + tid * 64, 0, 3);
    }

#pragma unroll
    for (int i = 0; i < 16; ++i) {
        int e = (tid + 256 * i) * 4;
        float4 vv = *(const float4*)(vws + row0 * 128 + e);
        float4 gv = *(const float4*)(ilf + row0 * 128 + e);
        int r = e >> 7, c = e & 127;
        unsigned short* p = &As[r * 144 + c];
        p[0] = f2bf(vv.x * gv.x); p[1] = f2bf(vv.y * gv.y);
        p[2] = f2bf(vv.z * gv.z); p[3] = f2bf(vv.w * gv.w);
    }
    __syncthreads();

    for (int job = tid >> 5; job < 64; job += 8) {
        const int js    = __builtin_amdgcn_readfirstlane(job);
        const int strip = js >> 3;
        const int ct    = js & 7;
        const int col   = ct * 16 + l15;
        const int lrow  = strip * 16 + l15;
        f32x8 acc = zero8();
#pragma unroll
        for (int kk = 0; kk < 4; ++kk) {
            int kb = kk * 32;
            Frag a, b;
            a.q[0] = *(const uint4*)(&As[lrow * 144 + kb + half * 16]);
            a.q[1] = *(const uint4*)(&As[lrow * 144 + kb + half * 16 + 8]);
            const uint4* pb = (const uint4*)(Mtb + col * 128 + kb + half * 16);
            b.q[0] = pb[0];
            b.q[1] = pb[1];
            acc = wmma_bf16(a, b, acc);
        }
        const float bpv = bp[col];
        const long  grow = row0 + strip * 16;
#pragma unroll
        for (int r = 0; r < 8; ++r) {
            int  orow = half ? (r + 8) : r;
            long addr = (grow + orow) * 128 + col;
            out[addr] += acc[r] + bpv;
        }
    }
}

// ---------------------------------------------------------------------------
// Launch
// ---------------------------------------------------------------------------
extern "C" void kernel_launch(void* const* d_in, const int* in_sizes, int n_in,
                              void* d_out, int out_size, void* d_ws, size_t ws_size,
                              hipStream_t stream) {
    const float* x   = (const float*)d_in[0];
    const float* ilf = (const float*)d_in[1];
    const float* Wq  = (const float*)d_in[2];
    const float* Wk  = (const float*)d_in[3];
    const float* Wv  = (const float*)d_in[4];
    const float* rsc = (const float*)d_in[5];
    const float* Wp  = (const float*)d_in[6];
    const float* bp  = (const float*)d_in[7];
    const float* pw1 = (const float*)d_in[8];
    const float* pw2 = (const float*)d_in[9];
    float* out = (float*)d_out;

    char* ws = (char*)d_ws;
    unsigned short* Wtq = (unsigned short*)(ws + 0);
    unsigned short* Wtk = (unsigned short*)(ws + 32768);
    unsigned short* Wtv = (unsigned short*)(ws + 65536);
    float*          Grm = (float*)(ws + 98304);            // 16384 f32
    float*          ssq = (float*)(ws + 163840);           // 512 f32
    float*          ssk = (float*)(ws + 165888);           // 512 f32
    unsigned short* Mt  = (unsigned short*)(ws + 167936);  // 65536 bf16
    size_t big = 524288;
    unsigned short* qws = (unsigned short*)(ws + big);                       // 64 MB
    unsigned short* kws = (unsigned short*)(ws + big + 67108864ull);         // 64 MB
    float*          vws = (float*)(ws + big + 134217728ull);                 // 128 MB

    k0_prep<<<260, 256, 0, stream>>>(Wq, Wk, Wv, Wtq, Wtk, Wtv, Grm);
    k1_proj<<<2048, 256, 0, stream>>>(x, Wtq, Wtk, Wtv, qws, kws, vws);
    k2_gram<<<2048, 256, 0, stream>>>(qws, kws, Grm, ssq, ssk);
    k3_attn<<<16, 256, 0, stream>>>(Grm, ssq, ssk, rsc, Wp, Mt);
    k4_conv<<<4096, 256, 0, stream>>>(vws, pw1, pw2, out);
    k5_out<<<2048, 256, 0, stream>>>(vws, ilf, Mt, bp, out);
}
